// ProximityViT_62500364091848
// MI455X (gfx1250) — matrix operand order repaired and checked by reading, not compiled, not run on hardware
//
#include <hip/hip_runtime.h>
#include <math.h>

// Problem constants (from reference): B=8, N=100, D=H=768, TH=24, T=577.
#define NB   8
#define NBOX 100
#define DIM  768
#define TOK  577
#define THW  24
#define ROWS (NB * NBOX)        // 800 pooled rows
#define HAB_COLS 1536           // [ha | hb]
#define KCHUNKS (DIM / 32)      // 24
#define NTILES  (HAB_COLS / 16) // 96
#define MTILES  (ROWS / 16)     // 50

typedef __attribute__((ext_vector_type(16))) __bf16 v16bf;
typedef __attribute__((ext_vector_type(8)))  float  v8f;

union BF16x16 {
    v16bf v;
    uint4 q[2];
    unsigned short s[16];
};

__device__ __forceinline__ unsigned short f32_to_bf16(float f) {
    unsigned int u = __float_as_uint(f);
    u += 0x7FFFu + ((u >> 16) & 1u);   // round-to-nearest-even
    return (unsigned short)(u >> 16);
}

// ---------------------------------------------------------------------------
// Kernel 1: ROI mean pooling -> pooled bf16 [800, 768]
// one block per (b, box); threads stride D (coalesced)
// ---------------------------------------------------------------------------
__global__ __launch_bounds__(256) void pool_kernel(
    const float* __restrict__ tok,      // [8,577,768]
    const float* __restrict__ boxes,    // [8,100,4]
    const int*   __restrict__ imgH,
    const int*   __restrict__ imgW,
    unsigned short* __restrict__ pooled // [800,768] bf16
) {
    const int bn = blockIdx.x;          // 0..799
    const int b  = bn / NBOX;
    const float fw = (float)(*imgW);
    const float fh = (float)(*imgH);

    const float* bx = boxes + (size_t)bn * 4;
    const float x1 = bx[0], y1 = bx[1], x2 = bx[2], y2 = bx[3];

    // to_patch: floor(floor(v*size)/size*TH), then clip
    int px1 = (int)floorf(floorf(x1 * fw) / fw * (float)THW);
    int py1 = (int)floorf(floorf(y1 * fh) / fh * (float)THW);
    int px2 = (int)floorf(floorf(x2 * fw) / fw * (float)THW);
    int py2 = (int)floorf(floorf(y2 * fh) / fh * (float)THW);
    px1 = min(max(px1, 0), THW - 1);
    py1 = min(max(py1, 0), THW - 1);
    px2 = min(max(px2, 1), THW);
    py2 = min(max(py2, 1), THW);
    if (px2 <= px1) px2 = px1 + 1;
    if (py2 <= py1) py2 = py1 + 1;

    const float inv = 1.0f / (float)((px2 - px1) * (py2 - py1));
    const float* base = tok + (size_t)b * TOK * DIM + (size_t)DIM; // skip CLS

    for (int d = threadIdx.x; d < DIM; d += 256) {
        float acc = 0.0f;
        for (int y = py1; y < py2; ++y) {
            const float* row = base + (size_t)(y * THW) * DIM + d;
            for (int x = px1; x < px2; ++x)
                acc += row[(size_t)x * DIM];
        }
        pooled[(size_t)bn * DIM + d] = f32_to_bf16(acc * inv);
    }
}

// ---------------------------------------------------------------------------
// Kernel 2a: cast + pack W1 into WMMA B-fragment layout (bf16).
//   Bcat[k, n] = W1[k + (n >= 768 ? 768 : 0), n & 767]
//   W1p[((nt*24 + kc)*32 + lane)*16 + h] = Bcat[kc*32 + 16*(lane>>4) + h,
//                                               nt*16 + (lane&15)]
// so the GEMM reads one contiguous 32B block per lane per K-chunk.
// ---------------------------------------------------------------------------
__global__ __launch_bounds__(256) void pack_w1_kernel(
    const float* __restrict__ W1, unsigned short* __restrict__ W1p)
{
    const int idx = blockIdx.x * 256 + threadIdx.x;
    if (idx >= NTILES * KCHUNKS * 32 * 16) return;
    const int h    = idx & 15;
    const int lane = (idx >> 4) & 31;
    const int rest = idx >> 9;
    const int kc   = rest % KCHUNKS;
    const int nt   = rest / KCHUNKS;
    const int g  = lane >> 4;
    const int ml = lane & 15;
    const int kk   = kc * 32 + 16 * g + h;
    const int nCol = nt * 16 + ml;
    const int row  = kk + ((nCol >= DIM) ? DIM : 0);
    const int col  = nCol & (DIM - 1);
    W1p[idx] = f32_to_bf16(W1[(size_t)row * DIM + col]);
}

// ---------------------------------------------------------------------------
// Kernel 2b: WMMA bf16 GEMM  hab[800,1536] = pooled[800,768] @ Bcat[768,1536]
// One wave32 per 16x16 output tile. 50 M-tiles x 96 N-tiles = 4800 waves.
// Inner loop: 2x b128 A-load + 2x b128 packed-B load + 1 WMMA, unrolled x4.
// ---------------------------------------------------------------------------
__global__ __launch_bounds__(128) void gemm_wmma_kernel(
    const unsigned short* __restrict__ Abf,  // pooled bf16 [800,768]
    const unsigned short* __restrict__ W1p,  // packed B fragments
    float* __restrict__ hab)                 // [800,1536] f32
{
    const int wave = blockIdx.x * 4 + (threadIdx.x >> 5);
    const int lane = threadIdx.x & 31;
    const int mt = wave % MTILES;
    const int nt = wave / MTILES;
    const int m0 = mt * 16;
    const int n0 = nt * 16;
    const int g  = lane >> 4;                // lane group (0/1)
    const int ml = lane & 15;

    // A fragment: lane group g holds K {16g..16g+15} for row m0+ml
    // (matches packed B order) -> two contiguous b128s.
    const unsigned short* ap = Abf + (size_t)(m0 + ml) * DIM + 16 * g;
    // Packed B: one contiguous 32B block per lane per K-chunk.
    const unsigned short* bp = W1p + ((size_t)nt * KCHUNKS * 32 + lane) * 16;

    v8f c = {};
    #pragma unroll 4
    for (int kc = 0; kc < KCHUNKS; ++kc) {
        BF16x16 a, bb;
        a.q[0]  = *reinterpret_cast<const uint4*>(ap);       // K = +16g .. +16g+7
        a.q[1]  = *reinterpret_cast<const uint4*>(ap + 8);   // K = +16g+8 .. +15
        bb.q[0] = *reinterpret_cast<const uint4*>(bp);
        bb.q[1] = *reinterpret_cast<const uint4*>(bp + 8);
        __builtin_prefetch(bp + 512, 0, 0);                  // next K-chunk B block
        c = __builtin_amdgcn_wmma_f32_16x16x32_bf16(
                false, a.v, false, bb.v, (short)0, c, false, false);
        ap += 32;
        bp += 512;
    }

    // D layout: VGPR r -> M = m0 + r + 8g, N = n0 + (lane&15)
    const int mBase = m0 + 8 * g;
    #pragma unroll
    for (int r = 0; r < 8; ++r)
        hab[(size_t)(mBase + r) * HAB_COLS + n0 + ml] = c[r];
}

// ---------------------------------------------------------------------------
// Kernel 3: fused pairwise relu-dot + sigmoid (never materializes [B,N,N,H])
// one block per (b, i); LDS caches ha_i + b1 and W2; one wave per j.
// ---------------------------------------------------------------------------
__global__ __launch_bounds__(256) void pairwise_kernel(
    const float* __restrict__ hab,   // [800,1536]: [ha | hb]
    const float* __restrict__ b1,    // [768]
    const float* __restrict__ W2,    // [768]
    const float* __restrict__ b2,    // [1]
    float* __restrict__ out)         // [8,100,100]
{
    __shared__ float sA[DIM];
    __shared__ float sW[DIM];
    const int bi = blockIdx.x;              // 0..799
    const int b  = bi / NBOX;
    const int i  = bi % NBOX;
    const int tid  = threadIdx.x;
    const int lane = tid & 31;
    const int wv   = tid >> 5;              // 0..7

    const float* ha = hab + (size_t)bi * HAB_COLS;
    for (int h = tid; h < DIM; h += 256) {
        sA[h] = ha[h] + b1[h];
        sW[h] = W2[h];
    }
    __syncthreads();

    const float bias2 = b2[0];
    for (int j0 = 0; j0 < NBOX; j0 += 8) {
        const int j = j0 + wv;
        if (j < NBOX) {
            const float* hb = hab + (size_t)(b * NBOX + j) * HAB_COLS + DIM;
            float sum = 0.0f;
            #pragma unroll
            for (int t = 0; t < DIM / 32; ++t) {
                const int h = lane + (t << 5);
                const float v = sA[h] + hb[h];
                sum = fmaf(fmaxf(v, 0.0f), sW[h], sum);
            }
            #pragma unroll
            for (int off = 16; off > 0; off >>= 1)
                sum += __shfl_down(sum, off, 32);
            if (lane == 0) {
                const float z = sum + bias2;
                out[((size_t)b * NBOX + i) * NBOX + j] = 1.0f / (1.0f + expf(-z));
            }
        }
    }
}

// ---------------------------------------------------------------------------
extern "C" void kernel_launch(void* const* d_in, const int* in_sizes, int n_in,
                              void* d_out, int out_size, void* d_ws, size_t ws_size,
                              hipStream_t stream) {
    const float* patch_tokens = (const float*)d_in[0];
    const float* boxes        = (const float*)d_in[1];
    const float* W1           = (const float*)d_in[2];
    const float* b1           = (const float*)d_in[3];
    const float* W2           = (const float*)d_in[4];
    const float* b2           = (const float*)d_in[5];
    const int*   img_h        = (const int*)d_in[6];
    const int*   img_w        = (const int*)d_in[7];
    float* out = (float*)d_out;

    // Workspace layout (16/32-byte aligned slices)
    const size_t pooled_bytes = (size_t)ROWS * DIM * sizeof(unsigned short);   // 1,228,800
    const size_t w1p_bytes    = (size_t)2 * DIM * DIM * sizeof(unsigned short);// 2,359,296
    unsigned short* pooled_bf = (unsigned short*)d_ws;
    unsigned short* w1_packed = (unsigned short*)((char*)d_ws + pooled_bytes);
    float*          hab       = (float*)((char*)d_ws + pooled_bytes + w1p_bytes);

    // 1. ROI mean-pool -> bf16 pooled rows
    pool_kernel<<<ROWS, 256, 0, stream>>>(patch_tokens, boxes, img_h, img_w, pooled_bf);

    // 2a. W1 -> bf16, packed into B-fragment layout
    const int w1n = 2 * DIM * DIM;
    pack_w1_kernel<<<(w1n + 255) / 256, 256, 0, stream>>>(W1, w1_packed);

    // 2b. WMMA GEMM: 4800 tiles, 4 waves per block
    gemm_wmma_kernel<<<(MTILES * NTILES) / 4, 128, 0, stream>>>(pooled_bf, w1_packed, hab);

    // 3. fused pairwise relu-dot + sigmoid
    pairwise_kernel<<<ROWS, 256, 0, stream>>>(hab, b1, W2, b2, out);
}
